// GraphAttentionLayer_53687091200589
// MI455X (gfx1250) — matrix operand order repaired and checked by reading, not compiled, not run on hardware
//
#include <hip/hip_runtime.h>
#include <hip/hip_bf16.h>

// ---------------------------------------------------------------------------
// GAT layer for MI455X (gfx1250, wave32, WMMA).
//   s1 = input @ (W @ a1)            (algebraic collapse of h_doc)
//   h_svc = serv_emb @ W  (split-bf16 hi/lo, stored TRANSPOSED [col][k])
//   flash-softmax over masked leaky_relu(s1_i + s2_k), attn @ h_svc via
//   3x v_wmma_f32_16x16x32_bf16; B tiles staged with async global->LDS copies.
// ---------------------------------------------------------------------------

typedef __bf16 bf16x16 __attribute__((ext_vector_type(16)));
typedef float  f32x8   __attribute__((ext_vector_type(8)));

union BFrag {
    bf16x16        v;
    unsigned short us[16];
    unsigned int   u[8];
    uint4          q[2];
};

__device__ __forceinline__ unsigned short f32_to_bf16_rne(float f) {
    unsigned int u = __float_as_uint(f);
    unsigned int r = u + 0x7FFFu + ((u >> 16) & 1u);
    return (unsigned short)(r >> 16);
}
__device__ __forceinline__ float bf16_bits_to_f32(unsigned short h) {
    return __uint_as_float(((unsigned int)h) << 16);
}

// Raw 32-bit LDS offset for async-copy VDST operand (generic -> AS3 -> int).
__device__ __forceinline__ unsigned lds_off(const void* p) {
    return (unsigned)(unsigned long long)(__attribute__((address_space(3))) const void*)p;
}

// Async 16-byte global -> LDS copy (ASYNCcnt-tracked, per CDNA5 ISA).
__device__ __forceinline__ void async_copy_b128(unsigned lds_dst, const void* gsrc) {
    unsigned long long ga = (unsigned long long)gsrc;
    asm volatile("global_load_async_to_lds_b128 %0, %1, off"
                 :: "v"(lds_dst), "v"(ga) : "memory");
}
__device__ __forceinline__ void wait_async0() {
    asm volatile("s_wait_asynccnt 0x0" ::: "memory");
}

#define GAT_N_DIM 128
#define GAT_K     2048
#define NEG_BIG   (-9.0e15f)
#define BPAD      40   // padded halves per b-tile row   (conflict-free b128)
#define APAD      20   // padded dwords per a-tile row   (conflict-free b128)

// ---------------------------------------------------------------------------
// Kernel 1: wa1[d] = sum_j W[d][j] * a[j]
// ---------------------------------------------------------------------------
__global__ __launch_bounds__(128) void gat_prep_wa1(const float* __restrict__ W,
                                                    const float* __restrict__ a,
                                                    float* __restrict__ wa1) {
    int d = threadIdx.x;
    float v = 0.f;
    #pragma unroll 8
    for (int j = 0; j < GAT_N_DIM; ++j)
        v += W[(size_t)d * GAT_N_DIM + j] * a[j];
    wa1[d] = v;
}

// ---------------------------------------------------------------------------
// Kernel 2: one block per service row k (128 threads):
//   h[j] = serv_emb[k] . W[:,j]  -> split bf16 hi/lo, stored transposed [j][k]
//   s2[k] = h . a2
// ---------------------------------------------------------------------------
__global__ __launch_bounds__(128) void gat_prep_svc(const float* __restrict__ serv_emb,
                                                    const float* __restrict__ W,
                                                    const float* __restrict__ a,
                                                    float* __restrict__ s2,
                                                    unsigned short* __restrict__ hsT_hi,
                                                    unsigned short* __restrict__ hsT_lo) {
    const int k = blockIdx.x;
    const int j = threadIdx.x;
    __shared__ float se[GAT_N_DIM];
    __shared__ float partial[4];

    se[j] = serv_emb[(size_t)k * GAT_N_DIM + j];
    __syncthreads();

    float h = 0.f;
    #pragma unroll 8
    for (int d = 0; d < GAT_N_DIM; ++d)
        h += se[d] * W[(size_t)d * GAT_N_DIM + j];

    unsigned short hi = f32_to_bf16_rne(h);
    float lo = h - bf16_bits_to_f32(hi);
    hsT_hi[(size_t)j * GAT_K + k] = hi;                 // transposed store
    hsT_lo[(size_t)j * GAT_K + k] = f32_to_bf16_rne(lo);

    float v = h * a[GAT_N_DIM + j];                     // a2 = a[128..255]
    #pragma unroll
    for (int m = 1; m < 32; m <<= 1) v += __shfl_xor(v, m, 32);
    if ((j & 31) == 0) partial[j >> 5] = v;
    __syncthreads();
    if (j == 0) s2[k] = partial[0] + partial[1] + partial[2] + partial[3];
}

// ---------------------------------------------------------------------------
// Kernel 3: main flash-attention GAT kernel.
// One block = 16 doc rows, 256 threads = 8 waves; wave w owns out cols w*16..
// ---------------------------------------------------------------------------
__global__ __launch_bounds__(256) void gat_main(const float* __restrict__ input,
                                                const int* __restrict__ adj,
                                                const float* __restrict__ wa1_g,
                                                const float* __restrict__ s2_g,
                                                const unsigned short* __restrict__ hsT_hi,
                                                const unsigned short* __restrict__ hsT_lo,
                                                float* __restrict__ out) {
    // B tiles transposed: [col][k], padded row stride -> frag = 2x ds_load_b128
    __shared__ __align__(16) unsigned short bT_hi[GAT_N_DIM][BPAD];  // 10 KB
    __shared__ __align__(16) unsigned short bT_lo[GAT_N_DIM][BPAD];  // 10 KB
    __shared__ __align__(16) unsigned int   a_hi[16][APAD];          // 1.25 KB
    __shared__ __align__(16) unsigned int   a_lo[16][APAD];
    __shared__ __align__(16) float s2s[GAT_K];                       // 8 KB
    __shared__ __align__(16) float wa1s[GAT_N_DIM];
    __shared__ float rowm[16], rowl[16], rowscale[16];

    const int t    = threadIdx.x;
    const int row0 = blockIdx.x * 16;

    {   // stage s2 + wa1 into LDS
        const float4* src = (const float4*)s2_g;
        float4*       dst = (float4*)s2s;
        for (int i = t; i < GAT_K / 4; i += 256) dst[i] = src[i];
        if (t < GAT_N_DIM / 4) ((float4*)wa1s)[t] = ((const float4*)wa1_g)[t];
        if (t < 16) { rowm[t] = NEG_BIG; rowl[t] = 0.f; }
    }
    __syncthreads();

    const int r = t >> 4;   // doc row within block (0..15)
    const int q = t & 15;   // 16 threads cooperate per row

    // s1 for this block's 16 rows: 16 threads x 8 elems each, shfl-reduce.
    float s1r;
    {
        const float* ip = input + (size_t)(row0 + r) * GAT_N_DIM + q * 8;
        const float* wp = wa1s + q * 8;
        float p = 0.f;
        #pragma unroll
        for (int e = 0; e < 8; ++e) p += ip[e] * wp[e];
        #pragma unroll
        for (int m = 1; m < 16; m <<= 1) p += __shfl_xor(p, m, 32);
        s1r = p;
    }

    const int wave   = t >> 5;        // 0..7 -> output column tile
    const int lane   = t & 31;
    const int laneHi = lane >> 4;
    const int ln     = lane & 15;
    const int col0   = wave * 16;

    const size_t adjBase = (size_t)(row0 + r) * GAT_K;

    // Async-staging assignment: 2 b128 segments per thread per tile.
    const int sc0 = (t)        >> 2, ss0 = (t) & 3;          // col, seg (idx 0..255)
    const int sc1 = (t + 256)  >> 2, ss1 = (t + 256) & 3;    // idx 256..511

    f32x8 acc = {0.f, 0.f, 0.f, 0.f, 0.f, 0.f, 0.f, 0.f};

    for (int k0 = 0; k0 < GAT_K; k0 += 32) {
        // ---- kick off async global->LDS staging of the h_svc chunk (hi+lo)
        async_copy_b128(lds_off(&bT_hi[sc0][ss0 * 8]), hsT_hi + (size_t)sc0 * GAT_K + k0 + ss0 * 8);
        async_copy_b128(lds_off(&bT_hi[sc1][ss1 * 8]), hsT_hi + (size_t)sc1 * GAT_K + k0 + ss1 * 8);
        async_copy_b128(lds_off(&bT_lo[sc0][ss0 * 8]), hsT_lo + (size_t)sc0 * GAT_K + k0 + ss0 * 8);
        async_copy_b128(lds_off(&bT_lo[sc1][ss1 * 8]), hsT_lo + (size_t)sc1 * GAT_K + k0 + ss1 * 8);

        // ---- masked leaky-relu logits: 2 per thread, coalesced adj stream
        if (k0 + 64 <= GAT_K)
            __builtin_prefetch(adj + adjBase + k0 + 32 + q * 2, 0, 1);

        int2  av = *(const int2*)(adj + adjBase + k0 + q * 2);
        float e0 = s1r + s2s[k0 + q * 2];
        float e1 = s1r + s2s[k0 + q * 2 + 1];
        e0 = (e0 > 0.f) ? e0 : 0.2f * e0;
        e1 = (e1 > 0.f) ? e1 : 0.2f * e1;
        e0 = (av.x > 0) ? e0 : NEG_BIG;
        e1 = (av.y > 0) ? e1 : NEG_BIG;

        // ---- online softmax: chunk max / sum via 16-lane xor shuffles
        float cm = fmaxf(e0, e1);
        #pragma unroll
        for (int m = 1; m < 16; m <<= 1) cm = fmaxf(cm, __shfl_xor(cm, m, 32));
        float m_old = rowm[r];
        float m_new = fmaxf(m_old, cm);
        float p0 = __expf(e0 - m_new);
        float p1 = __expf(e1 - m_new);
        float cs = p0 + p1;
        #pragma unroll
        for (int m = 1; m < 16; m <<= 1) cs += __shfl_xor(cs, m, 32);
        if (q == 0) {
            float sc    = __expf(m_old - m_new);   // finite: NEG_BIG is finite
            rowscale[r] = sc;
            rowl[r]     = rowl[r] * sc + cs;
            rowm[r]     = m_new;
        }

        // ---- split probabilities into bf16 hi/lo, pack k-pairs
        unsigned short h0 = f32_to_bf16_rne(p0);
        unsigned short h1 = f32_to_bf16_rne(p1);
        unsigned short l0 = f32_to_bf16_rne(p0 - bf16_bits_to_f32(h0));
        unsigned short l1 = f32_to_bf16_rne(p1 - bf16_bits_to_f32(h1));
        a_hi[r][q] = (unsigned)h0 | ((unsigned)h1 << 16);
        a_lo[r][q] = (unsigned)l0 | ((unsigned)l1 << 16);

        wait_async0();       // b tiles landed (this wave's copies)
        __syncthreads();     // all waves' copies + a tiles visible

        // ---- rescale accumulator by exp(m_old - m_new) per row
        #pragma unroll
        for (int jj = 0; jj < 8; ++jj) acc[jj] *= rowscale[jj + laneHi * 8];

        // ---- fragments: 2x ds_load_b128 each (padded, conflict-free)
        BFrag A, AL, B, BL;
        // A (16x32 bf16): lane ln -> M=ln; packed pairs are two contiguous runs.
        A.q[0]  = *(const uint4*)&a_hi[ln][laneHi * 4];
        A.q[1]  = *(const uint4*)&a_hi[ln][8 + laneHi * 4];
        AL.q[0] = *(const uint4*)&a_lo[ln][laneHi * 4];
        AL.q[1] = *(const uint4*)&a_lo[ln][8 + laneHi * 4];
        // B (32x16 bf16): lanes 0-15 K=0..7,16..23, lanes 16-31 K=8..15,24..31, N=col0+ln.
        B.q[0]  = *(const uint4*)&bT_hi[col0 + ln][laneHi * 8];
        B.q[1]  = *(const uint4*)&bT_hi[col0 + ln][16 + laneHi * 8];
        BL.q[0] = *(const uint4*)&bT_lo[col0 + ln][laneHi * 8];
        BL.q[1] = *(const uint4*)&bT_lo[col0 + ln][16 + laneHi * 8];

        // ---- 3-term error-compensated bf16 WMMA
        acc = __builtin_amdgcn_wmma_f32_16x16x32_bf16(false, A.v,  false, B.v,  (short)0, acc, false, false);
        acc = __builtin_amdgcn_wmma_f32_16x16x32_bf16(false, A.v,  false, BL.v, (short)0, acc, false, false);
        acc = __builtin_amdgcn_wmma_f32_16x16x32_bf16(false, AL.v, false, B.v,  (short)0, acc, false, false);
        __syncthreads();     // protect LDS tiles before restaging
    }

    // ---- epilogue: divide by softmax sum, ELU, store
    #pragma unroll
    for (int jj = 0; jj < 8; ++jj) {
        int   M = jj + laneHi * 8;
        float v = acc[jj] / rowl[M];
        v = (v > 0.f) ? v : (__expf(v) - 1.0f);
        out[(size_t)(row0 + M) * GAT_N_DIM + col0 + ln] = v;
    }
}

// ---------------------------------------------------------------------------
extern "C" void kernel_launch(void* const* d_in, const int* in_sizes, int n_in,
                              void* d_out, int out_size, void* d_ws, size_t ws_size,
                              hipStream_t stream) {
    const float* input = (const float*)d_in[0];   // N x 128
    const int*   adj   = (const int*)d_in[1];     // N x 2048
    const float* serv  = (const float*)d_in[2];   // 2048 x 128
    const float* W     = (const float*)d_in[3];   // 128 x 128
    const float* a     = (const float*)d_in[4];   // 256 x 1
    float*       out   = (float*)d_out;

    const int N = in_sizes[0] / GAT_N_DIM;        // 50000 (divisible by 16)

    // Workspace layout (~1.06 MB total)
    char* ws = (char*)d_ws;
    float*          wa1    = (float*)ws;                                  // 512 B
    float*          s2     = (float*)(ws + 512);                          // 8 KB
    unsigned short* hsT_hi = (unsigned short*)(ws + 512 + 8192);          // 512 KB
    unsigned short* hsT_lo = hsT_hi + (size_t)GAT_K * GAT_N_DIM;          // 512 KB

    gat_prep_wa1<<<1, 128, 0, stream>>>(W, a, wa1);
    gat_prep_svc<<<GAT_K, 128, 0, stream>>>(serv, W, a, s2, hsT_hi, hsT_lo);
    gat_main<<<N / 16, 256, 0, stream>>>(input, adj, wa1, s2, hsT_hi, hsT_lo, out);
}
